// INTagger_4973572129011
// MI455X (gfx1250) — matrix operand order repaired and checked by reading, not compiled, not run on hardware
//
#include <hip/hip_runtime.h>
#include <hip/hip_bf16.h>
#include <stdint.h>

// ---------------------------------------------------------------------------
// Interaction-Network tagger on gfx1250 (MI455X), mixed precision:
//  f32 precompute of layer-1 linear projections, f16 WMMA (f32 accumulate)
//  for the edge/object MLP GEMMs, f32 reductions.
//  Weights are pre-swizzled once into WMMA B-fragment lane layout in global
//  memory so each wave fetches its B operand as a single contiguous v16h.
// ---------------------------------------------------------------------------

typedef _Float16 half_t;
typedef __attribute__((ext_vector_type(16))) _Float16 v16h;
typedef __attribute__((ext_vector_type(8)))  float    v8f;

#define B_SZ   512
#define N_P    60
#define NV     5
#define P_F    20
#define S_F    14
#define HID    60
#define DE     20
#define DO     24
#define NCLS   2

// Swizzled-weight region offsets (in halves); each 64x64 layer = 8 tiles *
// 512 halves = 4096, each 64x32 layer = 4 tiles * 512 = 2048.
#define WSW_FR_W2   0
#define WSW_FR_W3   4096
#define WSW_PV_W2   6144
#define WSW_PV_W3   10240
#define WSW_FO_W1   12288
#define WSW_FO_W2   16384
#define WSW_FO_W3   20480
#define WSW_TOTAL   22528

// ---------------- WMMA helpers (16x16x32 f16 -> f32) -----------------------

__device__ __forceinline__ v8f wmma_f32_16x16x32(v16h a, v16h b, v8f c) {
  return __builtin_amdgcn_wmma_f32_16x16x32_f16(
      /*neg_a=*/false, a, /*neg_b=*/false, b,
      /*c_mod=*/(short)0, c, /*reuse_a=*/false, /*reuse_b=*/false);
}

// A fragment: 16x32 f16 from row-major LDS (ISA 7.12.2 16-bit A layout).
// Per lane this is two contiguous 16-byte runs -> 2x ds_load_b128.
__device__ __forceinline__ v16h load_frag_a(const half_t* lds, int row0,
                                            int stride, int kbase, int lane) {
  const int row = lane & 15, hi = lane >> 4;
  const half_t* p = lds + (row0 + row) * stride + kbase;
  v16h a;
#pragma unroll
  for (int v = 0; v < 8; ++v) {
    const int k0 = ((v < 4) ? 0 : 16) + hi * 8 + (v & 3) * 2;
    a[2 * v]     = p[k0];
    a[2 * v + 1] = p[k0 + 1];
  }
  return a;
}

// ---------------- Tiled MLP layers (A from LDS, B pre-swizzled global) -----
// out(f16)[64 x 16*ntiles] = relu(A[64x64] @ W + bias); 4 waves, wave m owns
// rows m*16..m*16+15. Wsw tile order: tileIdx = nt*2 + kt, 32 v16h per tile.

__device__ __forceinline__ void gemm64_relu_f16out(
    const half_t* A, const v16h* __restrict__ Wsw, const float* bias,
    int bias_n, half_t* out, int ostride, int ntiles, int tid) {
  const int wave = tid >> 5, lane = tid & 31;
  const int csel = lane & 15, hi = lane >> 4;
  const v16h a0 = load_frag_a(A, wave * 16, 64, 0, lane);
  const v16h a1 = load_frag_a(A, wave * 16, 64, 32, lane);
#pragma unroll
  for (int nt = 0; nt < ntiles; ++nt) {
    v8f acc = {};
    acc = wmma_f32_16x16x32(a0, Wsw[(nt * 2 + 0) * 32 + lane], acc);
    acc = wmma_f32_16x16x32(a1, Wsw[(nt * 2 + 1) * 32 + lane], acc);
    const int col = nt * 16 + csel;
    const float bv = (col < bias_n) ? bias[col] : 0.0f;
#pragma unroll
    for (int i = 0; i < 8; ++i) {
      const int r = wave * 16 + hi * 8 + i;
      float v = acc[i] + bv;
      v = v > 0.0f ? v : 0.0f;
      out[r * ostride + col] = (half_t)v;
    }
  }
}

__device__ __forceinline__ void gemm64_relu_f32out(
    const half_t* A, const v16h* __restrict__ Wsw, const float* bias,
    int bias_n, float* out, int ostride, int ntiles, int tid) {
  const int wave = tid >> 5, lane = tid & 31;
  const int csel = lane & 15, hi = lane >> 4;
  const v16h a0 = load_frag_a(A, wave * 16, 64, 0, lane);
  const v16h a1 = load_frag_a(A, wave * 16, 64, 32, lane);
#pragma unroll
  for (int nt = 0; nt < ntiles; ++nt) {
    v8f acc = {};
    acc = wmma_f32_16x16x32(a0, Wsw[(nt * 2 + 0) * 32 + lane], acc);
    acc = wmma_f32_16x16x32(a1, Wsw[(nt * 2 + 1) * 32 + lane], acc);
    const int col = nt * 16 + csel;
    const float bv = (col < bias_n) ? bias[col] : 0.0f;
#pragma unroll
    for (int i = 0; i < 8; ++i) {
      const int r = wave * 16 + hi * 8 + i;
      float v = acc[i] + bv;
      v = v > 0.0f ? v : 0.0f;
      out[r * ostride + col] = v;
    }
  }
}

// ---------------- Kernel 0: weight pre-swizzle (one-shot, tiny) ------------
// B-fragment layout: tile (nt,kt), lane: col = lane&15, hi = lane>>4;
// element e holds W[kt*32 + (e<8 ? 0 : 16) + hi*8 + (e&7)][nt*16 + col] (0-padded).
__device__ __forceinline__ half_t swz(const float* __restrict__ W, int K,
                                      int Ncols, int o) {
  const int tile = o >> 9;          // 512 halves per tile
  const int rem  = o & 511;
  const int lane = rem >> 4;
  const int e    = rem & 15;
  const int nt = tile >> 1, kt = tile & 1;
  const int col = lane & 15, hi = lane >> 4;
  const int k = kt * 32 + ((e < 8) ? 0 : 16) + hi * 8 + (e & 7);
  const int n = nt * 16 + col;
  const float val = (k < K && n < Ncols) ? W[k * Ncols + n] : 0.f;
  return (half_t)val;
}

__global__ __launch_bounds__(256) void swizzle_kernel(
    const float* __restrict__ fr_w2, const float* __restrict__ fr_w3,
    const float* __restrict__ pv_w2, const float* __restrict__ pv_w3,
    const float* __restrict__ fo_w1, const float* __restrict__ fo_w2,
    const float* __restrict__ fo_w3, half_t* __restrict__ wsw) {
  const int o = blockIdx.x * 256 + threadIdx.x;
  if      (o < WSW_FR_W3) wsw[o] = swz(fr_w2, HID, HID, o - WSW_FR_W2);
  else if (o < WSW_PV_W2) wsw[o] = swz(fr_w3, HID, DE,  o - WSW_FR_W3);
  else if (o < WSW_PV_W3) wsw[o] = swz(pv_w2, HID, HID, o - WSW_PV_W2);
  else if (o < WSW_FO_W1) wsw[o] = swz(pv_w3, HID, DE,  o - WSW_PV_W3);
  else if (o < WSW_FO_W2) wsw[o] = swz(fo_w1, HID, HID, o - WSW_FO_W1);
  else if (o < WSW_FO_W3) wsw[o] = swz(fo_w2, HID, HID, o - WSW_FO_W2);
  else if (o < WSW_TOTAL) wsw[o] = swz(fo_w3, HID, DO,  o - WSW_FO_W3);
}

// ---------------- Kernel 1: layer-1 linear projections (exact f32) ---------
__global__ __launch_bounds__(256) void precompute_kernel(
    const float* __restrict__ x, const float* __restrict__ y,
    const float* __restrict__ fr_w1, const float* __restrict__ pv_w1,
    float* __restrict__ R1, float* __restrict__ S1,
    float* __restrict__ K1, float* __restrict__ V1) {
  const int idx = blockIdx.x * 256 + threadIdx.x;
  const int total1 = B_SZ * N_P * HID;
  if (idx < total1) {
    const int n = idx % HID;
    const int row = idx / HID;     // b*60 + p
    const int b = row / N_P, p = row % N_P;
    float r1 = 0.f, s1 = 0.f, k1 = 0.f;
#pragma unroll
    for (int f = 0; f < P_F; ++f) {
      const float xv = x[(b * P_F + f) * N_P + p];  // xt[b,p,f]
      r1 += xv * fr_w1[f * HID + n];
      s1 += xv * fr_w1[(P_F + f) * HID + n];
      k1 += xv * pv_w1[f * HID + n];
    }
    R1[idx] = r1; S1[idx] = s1; K1[idx] = k1;
  } else {
    const int j = idx - total1;
    if (j < B_SZ * NV * HID) {
      const int n = j % HID;
      const int row = j / HID;     // b*5 + v
      const int b = row / NV, v = row % NV;
      float acc = 0.f;
#pragma unroll
      for (int s = 0; s < S_F; ++s) {
        acc += y[(b * S_F + s) * NV + v] * pv_w1[(P_F + s) * HID + n];
      }
      V1[j] = acc;
    }
  }
}

// ---------------- Kernel 2: particle-particle edges ------------------------
__global__ __launch_bounds__(128) void fr_edge_kernel(
    const float* __restrict__ R1, const float* __restrict__ S1,
    const float* __restrict__ fr_b1, const float* __restrict__ fr_b2,
    const float* __restrict__ fr_b3, const half_t* __restrict__ wsw,
    float* __restrict__ Ebar_pp) {
  __shared__ half_t sh_h1[64 * 64];
  __shared__ half_t sh_h2[64 * 64];
  __shared__ float  sh_h3[64 * 32];
  __shared__ float  sh_rb[64];
  __shared__ float  sh_red[2][32];

  const int wg = blockIdx.x;
  const int b = wg / N_P, r = wg % N_P;
  const int tid = threadIdx.x;

  const float* R1r = R1 + (b * N_P + r) * HID;
  if (tid < 64) sh_rb[tid] = (tid < HID) ? (R1r[tid] + fr_b1[tid]) : 0.f;
  __syncthreads();

  // h1[e][n] = relu(R1[r][n] + S1[s][n] + b1[n]); packed 2 halves per store.
  uint32_t* h1u = (uint32_t*)sh_h1;
  for (int idx = tid; idx < 64 * 32; idx += 128) {
    const int e = idx >> 5, n = (idx & 31) * 2;
    float v0 = 0.f, v1 = 0.f;
    if (e < N_P - 1 && n < HID) {
      const int s = (e < r) ? e : e + 1;
      const float2 sv = *(const float2*)(S1 + (b * N_P + s) * HID + n);
      v0 = sh_rb[n] + sv.x;     v0 = v0 > 0.f ? v0 : 0.f;
      v1 = sh_rb[n + 1] + sv.y; v1 = v1 > 0.f ? v1 : 0.f;
    }
    union { half_t h[2]; uint32_t u; } pk;
    pk.h[0] = (half_t)v0; pk.h[1] = (half_t)v1;
    h1u[idx] = pk.u;
  }
  __syncthreads();

  gemm64_relu_f16out(sh_h1, (const v16h*)(wsw + WSW_FR_W2), fr_b2, HID,
                     sh_h2, 64, 4, tid);
  __syncthreads();
  gemm64_relu_f32out(sh_h2, (const v16h*)(wsw + WSW_FR_W3), fr_b3, DE,
                     sh_h3, 32, 2, tid);
  __syncthreads();

  // Sum real edges only (padded rows carry relu(bias) junk).
  if (tid < 64) {
    const int col = tid & 31, h = tid >> 5;
    const int e0 = h ? 30 : 0, e1 = h ? (N_P - 1) : 30;
    float s = 0.f;
    for (int e = e0; e < e1; ++e) s += sh_h3[e * 32 + col];
    sh_red[h][col] = s;
  }
  __syncthreads();
  if (tid < DE) {
    Ebar_pp[(b * N_P + r) * DE + tid] = sh_red[0][tid] + sh_red[1][tid];
  }
}

// ---------------- Kernel 3: particle-vertex edges --------------------------
// One WG per (b, chunk of 12 receivers) -> 12*5 = 60 rows -> 64x64 tile.
__global__ __launch_bounds__(128) void pv_edge_kernel(
    const float* __restrict__ K1, const float* __restrict__ V1,
    const float* __restrict__ pv_b1, const float* __restrict__ pv_b2,
    const float* __restrict__ pv_b3, const half_t* __restrict__ wsw,
    float* __restrict__ Ebar_pv) {
  __shared__ half_t sh_h1[64 * 64];
  __shared__ half_t sh_h2[64 * 64];
  __shared__ float  sh_h3[64 * 32];

  const int wg = blockIdx.x;
  const int b = wg / 5, rc = wg % 5;  // receiver chunk of 12
  const int tid = threadIdx.x;

  uint32_t* h1u = (uint32_t*)sh_h1;
  for (int idx = tid; idx < 64 * 32; idx += 128) {
    const int rv = idx >> 5, n = (idx & 31) * 2;
    float v0 = 0.f, v1 = 0.f;
    if (rv < 60 && n < HID) {
      const int rloc = rv / NV, v = rv % NV;
      const int r = rc * 12 + rloc;
      const float2 kv = *(const float2*)(K1 + (b * N_P + r) * HID + n);
      const float2 vv = *(const float2*)(V1 + (b * NV + v) * HID + n);
      const float2 bb = *(const float2*)(pv_b1 + n);
      v0 = kv.x + vv.x + bb.x; v0 = v0 > 0.f ? v0 : 0.f;
      v1 = kv.y + vv.y + bb.y; v1 = v1 > 0.f ? v1 : 0.f;
    }
    union { half_t h[2]; uint32_t u; } pk;
    pk.h[0] = (half_t)v0; pk.h[1] = (half_t)v1;
    h1u[idx] = pk.u;
  }
  __syncthreads();

  gemm64_relu_f16out(sh_h1, (const v16h*)(wsw + WSW_PV_W2), pv_b2, HID,
                     sh_h2, 64, 4, tid);
  __syncthreads();
  gemm64_relu_f32out(sh_h2, (const v16h*)(wsw + WSW_PV_W3), pv_b3, DE,
                     sh_h3, 32, 2, tid);
  __syncthreads();

  for (int idx = tid; idx < 12 * DE; idx += 128) {
    const int rloc = idx / DE, n = idx % DE;
    const int r = rc * 12 + rloc;
    float s = 0.f;
#pragma unroll
    for (int v = 0; v < NV; ++v) s += sh_h3[(rloc * NV + v) * 32 + n];
    Ebar_pv[(b * N_P + r) * DE + n] = s;
  }
}

// ---------------- Kernel 4: object MLP + classifier ------------------------
__global__ __launch_bounds__(128) void fo_kernel(
    const float* __restrict__ x,
    const float* __restrict__ Ebar_pp, const float* __restrict__ Ebar_pv,
    const float* __restrict__ fo_b1, const float* __restrict__ fo_b2,
    const float* __restrict__ fo_b3, const half_t* __restrict__ wsw,
    const float* __restrict__ fc_w, const float* __restrict__ fc_b,
    float* __restrict__ out) {
  __shared__ half_t sh_c [64 * 64];
  __shared__ half_t sh_hA[64 * 64];
  __shared__ half_t sh_hB[64 * 64];
  __shared__ float  sh_h3[64 * 32];
  __shared__ float  sh_s [DO];

  const int b = blockIdx.x;
  const int tid = threadIdx.x;

  // C = [xt | Ebar_pp | Ebar_pv] : 60x60, packed 2 halves per store.
  uint32_t* cu = (uint32_t*)sh_c;
  for (int idx = tid; idx < 64 * 32; idx += 128) {
    const int row = idx >> 5, col = (idx & 31) * 2;
    float v0 = 0.f, v1 = 0.f;
    if (row < N_P && col < 60) {
      if (col < 20) {
        v0 = x[(b * P_F + col) * N_P + row];
        v1 = x[(b * P_F + col + 1) * N_P + row];
      } else if (col < 40) {
        const float2 t = *(const float2*)(Ebar_pp + (b * N_P + row) * DE + col - 20);
        v0 = t.x; v1 = t.y;
      } else {
        const float2 t = *(const float2*)(Ebar_pv + (b * N_P + row) * DE + col - 40);
        v0 = t.x; v1 = t.y;
      }
    }
    union { half_t h[2]; uint32_t u; } pk;
    pk.h[0] = (half_t)v0; pk.h[1] = (half_t)v1;
    cu[idx] = pk.u;
  }
  __syncthreads();

  gemm64_relu_f16out(sh_c,  (const v16h*)(wsw + WSW_FO_W1), fo_b1, HID,
                     sh_hA, 64, 4, tid);
  __syncthreads();
  gemm64_relu_f16out(sh_hA, (const v16h*)(wsw + WSW_FO_W2), fo_b2, HID,
                     sh_hB, 64, 4, tid);
  __syncthreads();
  gemm64_relu_f32out(sh_hB, (const v16h*)(wsw + WSW_FO_W3), fo_b3, DO,
                     sh_h3, 32, 2, tid);
  __syncthreads();

  if (tid < DO) {  // sum real rows only
    float s = 0.f;
    for (int row = 0; row < N_P; ++row) s += sh_h3[row * 32 + tid];
    sh_s[tid] = s;
  }
  __syncthreads();
  if (tid < NCLS) {
    float o = fc_b[tid];
#pragma unroll
    for (int n = 0; n < DO; ++n) o += sh_s[n] * fc_w[n * NCLS + tid];
    out[b * NCLS + tid] = o;
  }
}

// ---------------------------------------------------------------------------

extern "C" void kernel_launch(void* const* d_in, const int* in_sizes, int n_in,
                              void* d_out, int out_size, void* d_ws, size_t ws_size,
                              hipStream_t stream) {
  const float* x     = (const float*)d_in[0];
  const float* y     = (const float*)d_in[1];
  const float* fr_w1 = (const float*)d_in[2];
  const float* fr_b1 = (const float*)d_in[3];
  const float* fr_w2 = (const float*)d_in[4];
  const float* fr_b2 = (const float*)d_in[5];
  const float* fr_w3 = (const float*)d_in[6];
  const float* fr_b3 = (const float*)d_in[7];
  const float* pv_w1 = (const float*)d_in[8];
  const float* pv_b1 = (const float*)d_in[9];
  const float* pv_w2 = (const float*)d_in[10];
  const float* pv_b2 = (const float*)d_in[11];
  const float* pv_w3 = (const float*)d_in[12];
  const float* pv_b3 = (const float*)d_in[13];
  const float* fo_w1 = (const float*)d_in[14];
  const float* fo_b1 = (const float*)d_in[15];
  const float* fo_w2 = (const float*)d_in[16];
  const float* fo_b2 = (const float*)d_in[17];
  const float* fo_w3 = (const float*)d_in[18];
  const float* fo_b3 = (const float*)d_in[19];
  const float* fc_w  = (const float*)d_in[20];
  const float* fc_b  = (const float*)d_in[21];
  float* out = (float*)d_out;

  // Workspace layout (floats; all region sizes are multiples of 64 floats,
  // so the trailing half_t region stays 32-byte aligned for v16h loads).
  float* ws = (float*)d_ws;
  float* R1      = ws;                          // 512*60*60
  float* S1      = R1 + B_SZ * N_P * HID;       // 512*60*60
  float* K1      = S1 + B_SZ * N_P * HID;       // 512*60*60
  float* V1      = K1 + B_SZ * N_P * HID;       // 512*5*60
  float* Ebar_pp = V1 + B_SZ * NV * HID;        // 512*60*20
  float* Ebar_pv = Ebar_pp + B_SZ * N_P * DE;   // 512*60*20
  half_t* wsw    = (half_t*)(Ebar_pv + B_SZ * N_P * DE);  // 22528 halves

  swizzle_kernel<<<(WSW_TOTAL + 255) / 256, 256, 0, stream>>>(
      fr_w2, fr_w3, pv_w2, pv_w3, fo_w1, fo_w2, fo_w3, wsw);

  const int total = B_SZ * N_P * HID + B_SZ * NV * HID;
  precompute_kernel<<<(total + 255) / 256, 256, 0, stream>>>(
      x, y, fr_w1, pv_w1, R1, S1, K1, V1);

  fr_edge_kernel<<<B_SZ * N_P, 128, 0, stream>>>(
      R1, S1, fr_b1, fr_b2, fr_b3, wsw, Ebar_pp);

  pv_edge_kernel<<<B_SZ * 5, 128, 0, stream>>>(
      K1, V1, pv_b1, pv_b2, pv_b3, wsw, Ebar_pv);

  fo_kernel<<<B_SZ, 128, 0, stream>>>(
      x, Ebar_pp, Ebar_pv, fo_b1, fo_b2, fo_b3, wsw, fc_w, fc_b, out);
}
